// QGINWithPooling_42125039239794
// MI455X (gfx1250) — matrix operand order repaired and checked by reading, not compiled, not run on hardware
//
#include <hip/hip_runtime.h>
#include <hip/hip_bf16.h>

typedef __attribute__((ext_vector_type(16))) __bf16 v16bf;
typedef __attribute__((ext_vector_type(8)))  __bf16 v8bf;
typedef __attribute__((ext_vector_type(8)))  float  v8f;

union V16U { v16bf v; v8bf h[2]; };

// ---------------------------------------------------------------------------
// Pack a [K,256] f32 weight into bf16, pre-swizzled into WMMA B-fragment
// layout for v_wmma_f32_16x16x32_bf16:
//   B (32x16, K x N) per k-step: lane L holds column n = L%16,
//   K elements k = (L/16)*8 + e (e<8), 16 + (L/16)*8 + e-8 (e>=8), stored contiguously (32B/lane).
// Packed index: ((j*Kt + t)*32 + L)*16 + e   (j = n/16 tile, t = k/32 tile)
// ---------------------------------------------------------------------------
__global__ void pack_weight_bf16(const float* __restrict__ W,
                                 __bf16* __restrict__ P, int K) {
  const int NCOL = 256;
  int idx = blockIdx.x * blockDim.x + threadIdx.x;
  if (idx >= K * NCOL) return;
  int e    = idx & 15;
  int L    = (idx >> 4) & 31;
  int rest = idx >> 9;
  int Kt   = K >> 5;
  int t    = rest % Kt;
  int j    = rest / Kt;
  int k    = t * 32 + (e < 8 ? (L >> 4) * 8 + e : 16 + (L >> 4) * 8 + (e - 8));
  int n    = j * 16 + (L & 15);
  P[idx] = (__bf16)W[k * NCOL + n];
}

__global__ void zero_f32(float* __restrict__ p, int n) {
  int i = blockIdx.x * blockDim.x + threadIdx.x;
  if (i < n) p[i] = 0.f;
}

// ---------------------------------------------------------------------------
// Edge scatter-add: acc[dst[e], :] += x[src[e], :]   (F = 128 or 256)
// One thread per (edge, 4-float group): float4 gather + 4 f32 atomics.
// ---------------------------------------------------------------------------
__global__ void scatter_add(const float* __restrict__ x,
                            float* __restrict__ acc,
                            const int* __restrict__ src,
                            const int* __restrict__ dst,
                            int E, int F) {
  long tid = (long)blockIdx.x * blockDim.x + threadIdx.x;
  int gpe = F >> 2;
  long e = tid / gpe;
  int  g = (int)(tid % gpe);
  if (e >= E) return;
  long s = src[e], d = dst[e];
  const float4 v = *(const float4*)(x + s * F + g * 4);
  float* p = acc + d * F + g * 4;
  atomicAdd(p + 0, v.x);
  atomicAdd(p + 1, v.y);
  atomicAdd(p + 2, v.z);
  atomicAdd(p + 3, v.w);
}

// ---------------------------------------------------------------------------
// Fused GIN MLP: out = relu( relu(in*Wa + ba) * Wb + bb )
// 16 rows per block, 256 threads (8 wave32s). Intermediate 16x256 tile lives
// in LDS as bf16. Each wave computes 2 of the 16 column tiles per GEMM.
// A fragment (16x32 bf16): lane L -> row m = L%16,
//   k = t*32 + { half*8 + 0..7 , 16 + half*8 + 0..7 }  (half = L/16)
// POOL=true: accumulate colsum into pooled[] instead of storing x2.
// ---------------------------------------------------------------------------
template <int K_IN, bool POOL>
__global__ __launch_bounds__(256) void gin_mlp(
    const float* __restrict__ in,     // [N, K_IN]
    const __bf16* __restrict__ Wa,    // packed [K_IN x 256]
    const float* __restrict__ ba,
    const __bf16* __restrict__ Wb,    // packed [256 x 256]
    const float* __restrict__ bb,
    float* __restrict__ out0,         // [N,256] (x1)       when !POOL
    float* __restrict__ out1,         // [N,256] (h1 seed)  when !POOL
    float* __restrict__ pooled,       // [256]              when POOL
    int N) {
  constexpr int H = 256;
  constexpr int PAD = 8;
  __shared__ __bf16 A_lds[16 * (K_IN + PAD)];
  __shared__ __bf16 T_lds[16 * (H + PAD)];
  __shared__ float  colsum[H];

  const int tid  = threadIdx.x;
  const int row0 = blockIdx.x * 16;

  // Stage A rows (f32 -> bf16) into LDS
  for (int i = tid; i < 16 * K_IN; i += 256) {
    int m = i / K_IN, k = i % K_IN;
    int r = row0 + m;
    float v = (r < N) ? in[(long)r * K_IN + k] : 0.f;
    A_lds[m * (K_IN + PAD) + k] = (__bf16)v;
  }
  if (POOL && tid < H) colsum[tid] = 0.f;
  __syncthreads();

  const int wave = tid >> 5;
  const int lane = tid & 31;
  const int half = lane >> 4;
  const int mrow = lane & 15;

  // ---- GEMM1: [16 x K_IN] * [K_IN x 256] + ba, relu -> T_lds (bf16) ----
  #pragma unroll
  for (int jj = 0; jj < 2; ++jj) {
    int j = wave + jj * 8;
    v8f acc = {};
    #pragma unroll
    for (int t = 0; t < K_IN / 32; ++t) {
      const __bf16* arow = &A_lds[mrow * (K_IN + PAD) + t * 32];
      V16U a;
      a.h[0] = *(const v8bf*)(arow + half * 8);
      a.h[1] = *(const v8bf*)(arow + 16 + half * 8);
      const __bf16* bp = Wa + ((long)(j * (K_IN / 32) + t) * 32 + lane) * 16;
      v16bf b = *(const v16bf*)bp;
      acc = __builtin_amdgcn_wmma_f32_16x16x32_bf16(
          false, a.v, false, b, (short)0, acc, false, false);
    }
    int n = j * 16 + mrow;
    float bias = ba[n];
    #pragma unroll
    for (int r = 0; r < 8; ++r) {
      int m = r + half * 8;
      float v = fmaxf(acc[r] + bias, 0.f);
      T_lds[m * (H + PAD) + n] = (__bf16)v;
    }
  }
  __syncthreads();

  // ---- GEMM2: [16 x 256] * [256 x 256] + bb, relu -> out / colsum ----
  #pragma unroll
  for (int jj = 0; jj < 2; ++jj) {
    int j = wave + jj * 8;
    v8f acc = {};
    #pragma unroll
    for (int t = 0; t < H / 32; ++t) {
      const __bf16* arow = &T_lds[mrow * (H + PAD) + t * 32];
      V16U a;
      a.h[0] = *(const v8bf*)(arow + half * 8);
      a.h[1] = *(const v8bf*)(arow + 16 + half * 8);
      const __bf16* bp = Wb + ((long)(j * (H / 32) + t) * 32 + lane) * 16;
      v16bf b = *(const v16bf*)bp;
      acc = __builtin_amdgcn_wmma_f32_16x16x32_bf16(
          false, a.v, false, b, (short)0, acc, false, false);
    }
    int n = j * 16 + mrow;
    float bias = bb[n];
    float psum = 0.f;
    #pragma unroll
    for (int r = 0; r < 8; ++r) {
      int m = r + half * 8;
      float v = fmaxf(acc[r] + bias, 0.f);
      int rg = row0 + m;
      if (rg < N) {
        if (!POOL) {
          out0[(long)rg * H + n] = v;  // x1
          out1[(long)rg * H + n] = v;  // h1 seed (saves a 51MB memcpy pass)
        }
        psum += v;
      }
    }
    if (POOL) atomicAdd(&colsum[n], psum);
  }
  if (POOL) {
    __syncthreads();
    if (tid < H) atomicAdd(&pooled[tid], colsum[tid]);
  }
}

// out[c] = 2 * sum_h pooled[h] * out_w[h,c] + out_b[c]   (C = 8)
__global__ void final_out(const float* __restrict__ pooled,
                          const float* __restrict__ out_w,
                          const float* __restrict__ out_b,
                          float* __restrict__ out) {
  int c = threadIdx.x;
  if (c >= 8) return;
  float s = 0.f;
  for (int h = 0; h < 256; ++h) s += pooled[h] * out_w[h * 8 + c];
  out[c] = 2.f * s + out_b[c];
}

extern "C" void kernel_launch(void* const* d_in, const int* in_sizes, int n_in,
                              void* d_out, int out_size, void* d_ws,
                              size_t ws_size, hipStream_t stream) {
  const float* x   = (const float*)d_in[0];
  const int*   ei  = (const int*)d_in[1];
  const float* w0a = (const float*)d_in[4];
  const float* b0a = (const float*)d_in[5];
  const float* w0b = (const float*)d_in[6];
  const float* b0b = (const float*)d_in[7];
  const float* w1a = (const float*)d_in[8];
  const float* b1a = (const float*)d_in[9];
  const float* w1b = (const float*)d_in[10];
  const float* b1b = (const float*)d_in[11];
  const float* ow  = (const float*)d_in[12];
  const float* ob  = (const float*)d_in[13];

  const int FIN = 128, H = 256;
  const int N = in_sizes[0] / FIN;
  const int E = in_sizes[1] / 2;
  const int* src = ei;
  const int* dst = ei + E;

  char* ws = (char*)d_ws;
  size_t off = 0;
  auto alloc = [&](size_t bytes) -> char* {
    char* p = ws + off;
    off = (off + bytes + 255) & ~(size_t)255;
    return p;
  };
  __bf16* p0a    = (__bf16*)alloc((size_t)FIN * H * 2);
  __bf16* p0b    = (__bf16*)alloc((size_t)H * H * 2);
  __bf16* p1a    = (__bf16*)alloc((size_t)H * H * 2);
  __bf16* p1b    = (__bf16*)alloc((size_t)H * H * 2);
  float*  pooled = (float*)alloc((size_t)H * 4);
  float*  h0     = (float*)alloc((size_t)N * FIN * 4);
  float*  x1     = (float*)alloc((size_t)N * H * 4);
  float*  h1     = (float*)alloc((size_t)N * H * 4);

  // Weight packing (tiny; weights then stay L2-resident)
  pack_weight_bf16<<<(FIN * H + 255) / 256, 256, 0, stream>>>(w0a, p0a, FIN);
  pack_weight_bf16<<<(H * H + 255) / 256, 256, 0, stream>>>(w0b, p0b, H);
  pack_weight_bf16<<<(H * H + 255) / 256, 256, 0, stream>>>(w1a, p1a, H);
  pack_weight_bf16<<<(H * H + 255) / 256, 256, 0, stream>>>(w1b, p1b, H);
  zero_f32<<<1, 256, 0, stream>>>(pooled, H);

  // h0 = x, then h0 += scatter(x)
  hipMemcpyAsync(h0, x, (size_t)N * FIN * 4, hipMemcpyDeviceToDevice, stream);
  {
    long threads = (long)E * (FIN / 4);
    scatter_add<<<(unsigned)((threads + 255) / 256), 256, 0, stream>>>(
        x, h0, src, dst, E, FIN);
  }
  // x1 = MLP0(h0); also seed h1 = x1
  gin_mlp<128, false><<<(N + 15) / 16, 256, 0, stream>>>(
      h0, p0a, b0a, p0b, b0b, x1, h1, nullptr, N);
  // h1 += scatter(x1)
  {
    long threads = (long)E * (H / 4);
    scatter_add<<<(unsigned)((threads + 255) / 256), 256, 0, stream>>>(
        x1, h1, src, dst, E, H);
  }
  // x2 = MLP1(h1), reduced on the fly into pooled[]
  gin_mlp<256, true><<<(N + 15) / 16, 256, 0, stream>>>(
      h1, p1a, b1a, p1b, b1b, nullptr, nullptr, pooled, N);
  // out = 2*pooled @ out_w + out_b
  final_out<<<1, 32, 0, stream>>>(pooled, ow, ob, (float*)d_out);
}
